// CausalSelfAttention_42004780155695
// MI455X (gfx1250) — compile-verified, matrix operand order checked by reading, not run on hardware
//
#include <hip/hip_runtime.h>

#define BB 4
#define TT 2048
#define CC 2048
#define HH 16
#define DD 128

typedef __attribute__((ext_vector_type(16))) __bf16 v16bf;
typedef __attribute__((ext_vector_type(8)))  __bf16 v8bf;
typedef __attribute__((ext_vector_type(8)))  float  v8f;

static __device__ __forceinline__ v8f wmma_bf16(v16bf a, v16bf b, v8f c) {
  // D = A(16x32 bf16) * B(32x16 bf16) + C(16x16 f32)
  return __builtin_amdgcn_wmma_f32_16x16x32_bf16(false, a, false, b, (short)0, c,
                                                 false, false);
}

// Per-lane fragment: 8 contiguous bf16 at p0, 8 contiguous bf16 at p1.
// Matches ISA 16-bit A/B layout: lanes 0-15 K=[0..7],[16..23]; lanes 16-31
// K=[8..15],[24..31] when callers offset p0 by (lane>=16 ? 8 : 0).
static __device__ __forceinline__ v16bf load_frag(const __bf16* p0, const __bf16* p1) {
  v8bf lo = *(const v8bf*)p0;
  v8bf hi = *(const v8bf*)p1;
  v16bf r;
#pragma unroll
  for (int i = 0; i < 8; ++i) { r[i] = lo[i]; r[i + 8] = hi[i]; }
  return r;
}

static __device__ __forceinline__ v8f zero8() {
  v8f z = {0.f, 0.f, 0.f, 0.f, 0.f, 0.f, 0.f, 0.f};
  return z;
}

// ---- CDNA5 async global->LDS (ASYNCcnt-tracked, bypasses VGPRs) ------------
// ROCm builtin signature (from diagnostics): params are int4* in AS1 / AS3.
typedef int v4i __attribute__((vector_size(16)));
typedef __attribute__((address_space(1))) v4i* as1_v4i;
typedef __attribute__((address_space(3))) v4i* as3_v4i;

static __device__ __forceinline__ void async_copy16(const __bf16* g, __bf16* l) {
#if __has_builtin(__builtin_amdgcn_global_load_async_to_lds_b128)
  __builtin_amdgcn_global_load_async_to_lds_b128(
      (as1_v4i)(v4i*)g, (as3_v4i)(v4i*)l, 0, 0);
#else
  *(v8bf*)l = *(const v8bf*)g;
#endif
}

static __device__ __forceinline__ void wait_async0() {
#if __has_builtin(__builtin_amdgcn_s_wait_asynccnt)
  __builtin_amdgcn_s_wait_asynccnt(0);
#else
  asm volatile("s_wait_asynccnt 0" ::: "memory");
#endif
}

// ---------------------------------------------------------------------------
// Kernel 1: fp32 -> bf16 conversion, 8 elements/thread (b128 traffic)
// ---------------------------------------------------------------------------
__global__ void f32_to_bf16(const float* __restrict__ in, __bf16* __restrict__ out,
                            int n) {
  int i = (blockIdx.x * blockDim.x + threadIdx.x) * 8;
  if (i < n) {
    v8bf o;
#pragma unroll
    for (int j = 0; j < 8; ++j) o[j] = (__bf16)in[i + j];
    *(v8bf*)(out + i) = o;
  }
}

// ---------------------------------------------------------------------------
// Kernel 2/5: C[M,N] = A[M,K] * B[N,K]^T + bias   (bf16 in, f32 acc)
// Block tile 128x128, BK=32, double-buffered LDS fed by async loads.
// 8 waves (4 along M x 2 along N), each wave 2x4 WMMA tiles of 16x16.
// ---------------------------------------------------------------------------
template <typename OutT>
__global__ __launch_bounds__(256) void gemm_bf16_nt(
    const __bf16* __restrict__ A, const __bf16* __restrict__ Bm,
    const float* __restrict__ bias, OutT* __restrict__ outp,
    int M, int N, int Kdim) {
  __shared__ __bf16 sA[2][128][40];  // 128x32 tiles, padded rows (16B aligned)
  __shared__ __bf16 sB[2][128][40];

  const int tid = threadIdx.x;
  const int lane = tid & 31, wave = tid >> 5;
  const int l15 = lane & 15, lhalf = lane >> 4;
  const int wm = wave & 3, wn = wave >> 2;
  const int m0 = blockIdx.y * 128, n0 = blockIdx.x * 128;

  v8f acc[2][4];
#pragma unroll
  for (int i = 0; i < 2; ++i)
#pragma unroll
    for (int j = 0; j < 4; ++j) acc[i][j] = zero8();

  const int lr = tid >> 1;        // 0..127 : tile row
  const int lc = (tid & 1) * 16;  // 0 or 16

  const __bf16* gA = A + (size_t)(m0 + lr) * Kdim + lc;
  const __bf16* gB = Bm + (size_t)(n0 + lr) * Kdim + lc;

#define GEMM_LOAD_TILE(k0, buf)                         \
  do {                                                  \
    async_copy16(gA + (k0), &sA[buf][lr][lc]);          \
    async_copy16(gA + (k0) + 8, &sA[buf][lr][lc + 8]);  \
    async_copy16(gB + (k0), &sB[buf][lr][lc]);          \
    async_copy16(gB + (k0) + 8, &sB[buf][lr][lc + 8]);  \
  } while (0)

  GEMM_LOAD_TILE(0, 0);
  wait_async0();
  __syncthreads();

  const int nk = Kdim >> 5;
  for (int ik = 0; ik < nk; ++ik) {
    const int cur = ik & 1;
    if (ik + 1 < nk) GEMM_LOAD_TILE((ik + 1) * 32, cur ^ 1);

    const int ko = lhalf * 8;
    v16bf af[2], bfr[4];
#pragma unroll
    for (int mt = 0; mt < 2; ++mt) {
      const __bf16* p = &sA[cur][wm * 32 + mt * 16 + l15][ko];
      af[mt] = load_frag(p, p + 16);
    }
#pragma unroll
    for (int nt = 0; nt < 4; ++nt) {
      const __bf16* p = &sB[cur][wn * 64 + nt * 16 + l15][ko];
      bfr[nt] = load_frag(p, p + 16);
    }
#pragma unroll
    for (int mt = 0; mt < 2; ++mt)
#pragma unroll
      for (int nt = 0; nt < 4; ++nt)
        acc[mt][nt] = wmma_bf16(af[mt], bfr[nt], acc[mt][nt]);

    wait_async0();
    __syncthreads();
  }
#undef GEMM_LOAD_TILE

  // branch-free epilogue
  float bv[4];
#pragma unroll
  for (int nt = 0; nt < 4; ++nt) bv[nt] = bias[n0 + wn * 64 + nt * 16 + l15];

#pragma unroll
  for (int mt = 0; mt < 2; ++mt) {
    OutT* orow = outp + (size_t)(m0 + wm * 32 + mt * 16 + lhalf * 8) * N +
                 (n0 + wn * 64 + l15);
#pragma unroll
    for (int i = 0; i < 8; ++i) {
#pragma unroll
      for (int nt = 0; nt < 4; ++nt)
        orow[(size_t)i * N + nt * 16] = (OutT)(acc[mt][nt][i] + bv[nt]);
    }
  }
}

// ---------------------------------------------------------------------------
// Kernel 3: RoPE + split qkv[B*T,3C] -> q[b,h,t,d] (pre-scaled by 1/sqrt(D)),
// k[b,h,t,d], vT[b,h,d,t]
// reference: rot[j] = -x[2j+1] (j<64) ; rot[j] = x[2(j-64)] (j>=64)
//            out = x*(cos+sin) + rot*(cos-sin),  cos/sin at t*invf[d%64]
// ---------------------------------------------------------------------------
__global__ __launch_bounds__(256) void rope_split(
    const __bf16* __restrict__ qkv, __bf16* __restrict__ Qr,
    __bf16* __restrict__ Kr, __bf16* __restrict__ Vt) {
  size_t idx = (size_t)blockIdx.x * blockDim.x + threadIdx.x;  // ((b*H+h)*T+t)*D+d
  int d = (int)(idx & (DD - 1));
  size_t r = idx >> 7;
  int t = (int)(r & (TT - 1));
  r >>= 11;
  int h = (int)(r & (HH - 1));
  int b = (int)(r >> 4);

  const __bf16* row = qkv + ((size_t)b * TT + t) * (3 * CC);
  int col = h * DD + d;
  float qv = (float)row[col];
  float kv = (float)row[CC + col];
  float vv = (float)row[2 * CC + col];

  int fi = d & 63;
  int partner = (d < 64) ? (2 * d + 1) : (2 * (d - 64));
  float sgn = (d < 64) ? -1.f : 1.f;
  float qp = (float)row[h * DD + partner] * sgn;
  float kp = (float)row[CC + h * DD + partner] * sgn;

  // inv_freq = 10000^(-2*fi/128) = exp(-ln(1e4)/64 * fi)
  float ang = (float)t * __expf(-0.14391156f * (float)fi);
  float c = __cosf(ang), s = __sinf(ang);
  float cps = c + s, cms = c - s;

  const float SCALE = 0.0883883476483184f;  // 1/sqrt(128), folded into q
  Qr[idx] = (__bf16)((qv * cps + qp * cms) * SCALE);
  Kr[idx] = (__bf16)(kv * cps + kp * cms);
  Vt[(((size_t)b * HH + h) * DD + d) * TT + t] = (__bf16)vv;
}

// ---------------------------------------------------------------------------
// Kernel 4: causal flash attention, double-buffered async K/V tiles.
// grid = (T/128, B*H). 8 waves/block, 16 query rows per wave.
// Per 32-key tile: S=Q*K^T (8 wmma), online softmax, P*V (8 wmma).
// Wave-uniform skip of fully-masked tiles; masks only on the diagonal band.
// ---------------------------------------------------------------------------
__global__ __launch_bounds__(256) void flash_attn(
    const __bf16* __restrict__ Qr, const __bf16* __restrict__ Kr,
    const __bf16* __restrict__ Vt, __bf16* __restrict__ Y) {
  __shared__ __bf16 sK[2][32][136];    // 32 keys x 128 d (padded)
  __shared__ __bf16 sV[2][128][40];    // 128 d x 32 keys (padded)
  __shared__ __bf16 sP[8][16][40];     // per-wave P re-layout patch

  const int tid = threadIdx.x;
  const int lane = tid & 31, wave = tid >> 5;
  const int l15 = lane & 15, lhalf = lane >> 4;
  const int bh = blockIdx.y;
  const int q0 = blockIdx.x * 128;
  const int b = bh >> 4, h = bh & 15;
  const int qbase = q0 + wave * 16;  // this wave's first query row

  const __bf16* Q = Qr + (size_t)bh * TT * DD;
  const __bf16* K = Kr + (size_t)bh * TT * DD;
  const __bf16* V = Vt + (size_t)bh * DD * TT;

  // Q fragments for this wave's 16 rows (K-dim = D = 4 chunks of 32)
  const int qrow = qbase + l15;
  v16bf qf[4];
#pragma unroll
  for (int f = 0; f < 4; ++f) {
    const __bf16* p = Q + (size_t)qrow * DD + f * 32 + lhalf * 8;
    qf[f] = load_frag(p, p + 16);
  }

  v8f O[8];  // 16x128 f32 accumulator (8 d-tiles)
  float m_[8], l_[8];
#pragma unroll
  for (int i = 0; i < 8; ++i) { O[i] = zero8(); m_[i] = -3.0e38f; l_[i] = 0.f; }

  const int krow = tid >> 3, ks = (tid & 7) * 16;  // K tile 32x128
  const int vrow = tid >> 1, vs = (tid & 1) * 16;  // V^T tile 128x32

#define FA_LOAD_TILE(kt, buf)                                      \
  do {                                                             \
    const __bf16* gk = K + (size_t)((kt) + krow) * DD + ks;        \
    async_copy16(gk, &sK[buf][krow][ks]);                          \
    async_copy16(gk + 8, &sK[buf][krow][ks + 8]);                  \
    const __bf16* gv = V + (size_t)vrow * TT + (kt) + vs;          \
    async_copy16(gv, &sV[buf][vrow][vs]);                          \
    async_copy16(gv + 8, &sV[buf][vrow][vs + 8]);                  \
  } while (0)

  FA_LOAD_TILE(0, 0);
  wait_async0();
  __syncthreads();

  const int nkt = (q0 + 128) >> 5;
  for (int ik = 0; ik < nkt; ++ik) {
    const int cur = ik & 1;
    const int kt = ik * 32;
    if (ik + 1 < nkt) FA_LOAD_TILE(kt + 32, cur ^ 1);

    // Wave-uniform: skip tiles entirely above this wave's diagonal
    // (EXEC stays all-ones inside, as WMMA requires).
    if (kt <= qbase + 15) {
      const int ko = lhalf * 8;
      v8f S0 = zero8(), S1 = zero8();  // keys kt..kt+15 / kt+16..kt+31
#pragma unroll
      for (int f = 0; f < 4; ++f) {
        const __bf16* p0 = &sK[cur][l15][f * 32 + ko];
        const __bf16* p1 = &sK[cur][16 + l15][f * 32 + ko];
        v16bf kb0 = load_frag(p0, p0 + 16);
        v16bf kb1 = load_frag(p1, p1 + 16);
        S0 = wmma_bf16(qf[f], kb0, S0);
        S1 = wmma_bf16(qf[f], kb1, S1);
      }

      // masks only needed on the diagonal band (wave-uniform test)
      const bool need_mask = (kt + 31 > qbase);
      float alpha[8];
#pragma unroll
      for (int i = 0; i < 8; ++i) {
        float s0 = S0[i];
        float s1 = S1[i];
        if (need_mask) {
          int qg = qbase + i + lhalf * 8;
          if (kt + l15 > qg)      s0 = -1.0e30f;
          if (kt + 16 + l15 > qg) s1 = -1.0e30f;
        }
        float rm = fmaxf(s0, s1);
        rm = fmaxf(rm, __shfl_xor(rm, 1));
        rm = fmaxf(rm, __shfl_xor(rm, 2));
        rm = fmaxf(rm, __shfl_xor(rm, 4));
        rm = fmaxf(rm, __shfl_xor(rm, 8));
        float mn = fmaxf(m_[i], rm);
        float a = __expf(m_[i] - mn);
        m_[i] = mn;
        float p0 = __expf(s0 - mn);
        float p1 = __expf(s1 - mn);
        float rs = p0 + p1;
        rs += __shfl_xor(rs, 1);
        rs += __shfl_xor(rs, 2);
        rs += __shfl_xor(rs, 4);
        rs += __shfl_xor(rs, 8);
        l_[i] = l_[i] * a + rs;
        alpha[i] = a;
        // C-layout -> LDS (row M = i + lhalf*8, col N = key-in-tile)
        sP[wave][i + lhalf * 8][l15]      = (__bf16)p0;
        sP[wave][i + lhalf * 8][16 + l15] = (__bf16)p1;
      }
#pragma unroll
      for (int dt = 0; dt < 8; ++dt)
#pragma unroll
        for (int i = 0; i < 8; ++i) O[dt][i] *= alpha[i];

      // ensure our wave's P stores are visible before A-layout reload
      asm volatile("s_wait_dscnt 0" ::: "memory");
      const __bf16* pp = &sP[wave][l15][lhalf * 8];
      v16bf pf = load_frag(pp, pp + 16);
#pragma unroll
      for (int dt = 0; dt < 8; ++dt) {
        const __bf16* vp = &sV[cur][dt * 16 + l15][ko];
        v16bf vb = load_frag(vp, vp + 16);
        O[dt] = wmma_bf16(pf, vb, O[dt]);
      }
    }

    wait_async0();
    __syncthreads();
  }
#undef FA_LOAD_TILE

  // normalize and write y[b*T+row, h*D + d]
#pragma unroll
  for (int i = 0; i < 8; ++i) {
    float inv = 1.0f / l_[i];
    int row = qbase + i + lhalf * 8;
    __bf16* yrow = Y + ((size_t)b * TT + row) * CC + h * DD + l15;
#pragma unroll
    for (int dt = 0; dt < 8; ++dt)
      yrow[dt * 16] = (__bf16)(O[dt][i] * inv);
  }
}

// ---------------------------------------------------------------------------
// Launcher
// ---------------------------------------------------------------------------
extern "C" void kernel_launch(void* const* d_in, const int* in_sizes, int n_in,
                              void* d_out, int out_size, void* d_ws, size_t ws_size,
                              hipStream_t stream) {
  (void)in_sizes; (void)n_in; (void)out_size; (void)ws_size;
  const float* x      = (const float*)d_in[0];
  const float* w_attn = (const float*)d_in[1];
  const float* b_attn = (const float*)d_in[2];
  const float* w_proj = (const float*)d_in[3];
  const float* b_proj = (const float*)d_in[4];
  float* out = (float*)d_out;

  // workspace carve (bytes)
  char* ws = (char*)d_ws;
  __bf16* xb  = (__bf16*)(ws);                 //  33,554,432 B : x bf16
  __bf16* wab = (__bf16*)(ws + 33554432ull);   //  25,165,824 B : w_attn bf16
  __bf16* wpb = (__bf16*)(ws + 58720256ull);   //   8,388,608 B : w_proj bf16
  __bf16* qkv = (__bf16*)(ws + 67108864ull);   // 100,663,296 B : qkv bf16
  __bf16* qr  = (__bf16*)(ws + 167772160ull);  //  33,554,432 B : q roped*scale
  __bf16* kr  = (__bf16*)(ws + 201326592ull);  //  33,554,432 B : k roped
  __bf16* vt  = (__bf16*)(ws + 234881024ull);  //  33,554,432 B : v transposed
  __bf16* yb  = (__bf16*)(ws + 268435456ull);  //  33,554,432 B : attn out bf16

  const int nX  = BB * TT * CC;   // 16,777,216
  const int nWA = 3 * CC * CC;    // 12,582,912
  const int nWP = CC * CC;        //  4,194,304

  f32_to_bf16<<<nX / 2048, 256, 0, stream>>>(x, xb, nX);
  f32_to_bf16<<<nWA / 2048, 256, 0, stream>>>(w_attn, wab, nWA);
  f32_to_bf16<<<nWP / 2048, 256, 0, stream>>>(w_proj, wpb, nWP);

  // qkv = x @ w_attn^T + b_attn   (M=8192, N=6144, K=2048) -> bf16
  gemm_bf16_nt<__bf16><<<dim3(3 * CC / 128, BB * TT / 128), 256, 0, stream>>>(
      xb, wab, b_attn, qkv, BB * TT, 3 * CC, CC);

  // RoPE + split/transposition
  rope_split<<<(BB * HH * TT * DD) / 256, 256, 0, stream>>>(qkv, qr, kr, vt);

  // flash attention
  flash_attn<<<dim3(TT / 128, BB * HH), 256, 0, stream>>>(qr, kr, vt, yb);

  // out = y @ w_proj^T + b_proj   (M=8192, N=2048, K=2048) -> f32
  gemm_bf16_nt<float><<<dim3(CC / 128, BB * TT / 128), 256, 0, stream>>>(
      yb, wpb, b_proj, out, BB * TT, CC, CC);
}
